// CausalSelfAttention_31610959298969
// MI455X (gfx1250) — compile-verified, matrix operand order checked by reading, not run on hardware
//
#include <hip/hip_runtime.h>

#define DIM 2048
#define HEADS 16
#define BB 4
#define TT 2048
#define CHUNK 128
#define E 128      // head dim
#define DK 256     // lrpe'd q/k dim

typedef __attribute__((ext_vector_type(16))) __bf16 v16bf;
typedef __attribute__((ext_vector_type(8)))  float  v8f;
typedef unsigned short u16;

#define WMMA_BF16(A, B, C) \
  __builtin_amdgcn_wmma_f32_16x16x32_bf16(false, (A), false, (B), (short)0, (C), false, false)

static __device__ __forceinline__ u16 f2bf_bits(float f) {
  union { float f; unsigned u; } a; a.f = f;
  unsigned u = a.u;
  unsigned r = u + 0x7FFFu + ((u >> 16) & 1u);   // round-to-nearest-even
  return (u16)(r >> 16);
}
static __device__ __forceinline__ float bf2f(u16 h) {
  union { unsigned u; float f; } a; a.u = ((unsigned)h) << 16;
  return a.f;
}

// Load a 16x32 bf16 fragment (A-layout) or a 32x16 fragment stored as
// 16 rows x contiguous-K (B-layout): lane&15 selects row/col, lane>>4 selects
// which half of K. One 32-byte contiguous load per lane.
static __device__ __forceinline__ v16bf frag_ld(const u16* base, int ld, int row, int kk) {
  int l = threadIdx.x & 31;
  return *(const v16bf*)(base + (size_t)(row + (l & 15)) * (size_t)ld + kk + ((l >> 4) << 4));
}

static __device__ __forceinline__ void wmma8(const v16bf& a0, const v16bf& a1,
                                             const v16bf& b0, const v16bf& b1,
                                             const v16bf& b2, const v16bf& b3,
                                             v8f acc[8]) {
  acc[0] = WMMA_BF16(a0, b0, acc[0]);
  acc[1] = WMMA_BF16(a0, b1, acc[1]);
  acc[2] = WMMA_BF16(a0, b2, acc[2]);
  acc[3] = WMMA_BF16(a0, b3, acc[3]);
  acc[4] = WMMA_BF16(a1, b0, acc[4]);
  acc[5] = WMMA_BF16(a1, b1, acc[5]);
  acc[6] = WMMA_BF16(a1, b2, acc[6]);
  acc[7] = WMMA_BF16(a1, b3, acc[7]);
}

// ---------------------------------------------------------------- cast fp32 -> bf16 (x4)
__global__ void cast_bf16_kernel(const float* __restrict__ src, u16* __restrict__ dst, long n4) {
  long i = (long)blockIdx.x * blockDim.x + threadIdx.x;
  long stride = (long)gridDim.x * blockDim.x;
  const float4* s4 = (const float4*)src;
  unsigned long long* d4 = (unsigned long long*)dst;
  for (; i < n4; i += stride) {
    float4 f = s4[i];
    union { u16 h[4]; unsigned long long ull; } o;
    o.h[0] = f2bf_bits(f.x);
    o.h[1] = f2bf_bits(f.y);
    o.h[2] = f2bf_bits(f.z);
    o.h[3] = f2bf_bits(f.w);
    d4[i] = o.ull;
  }
}

// ---------------------------------------------------------------- QKV GEMM + fused LRPE
// qkv = x(8192x2048) @ w_qkv^T(2048x6144); epilogue applies LRPE to q,k and scatters
// q,k as (b,h,T,256) bf16 and v as (b,h,T,128) bf16.
__global__ __launch_bounds__(256) void qkv_lrpe_kernel(
    const u16* __restrict__ xb, const u16* __restrict__ wb,
    const float* __restrict__ theta,
    u16* __restrict__ qout, u16* __restrict__ kout, u16* __restrict__ vout) {
  int tid = threadIdx.x;
  int w = tid >> 5, l = tid & 31;
  int wr = w >> 1, wc = w & 1;
  int r0 = blockIdx.x * 128 + wr * 32;
  int c0 = blockIdx.y * 128 + wc * 64;

  v8f acc[8];
  v8f z = {};
  for (int i = 0; i < 8; i++) acc[i] = z;

  for (int kk = 0; kk < DIM; kk += 32) {
    v16bf a0 = frag_ld(xb, DIM, r0, kk);
    v16bf a1 = frag_ld(xb, DIM, r0 + 16, kk);
    v16bf b0 = frag_ld(wb, DIM, c0, kk);
    v16bf b1 = frag_ld(wb, DIM, c0 + 16, kk);
    v16bf b2 = frag_ld(wb, DIM, c0 + 32, kk);
    v16bf b3 = frag_ld(wb, DIM, c0 + 48, kk);
    wmma8(a0, a1, b0, b1, b2, b3, acc);
  }

  for (int ti = 0; ti < 2; ti++) {
    for (int tj = 0; tj < 4; tj++) {
      int mb = r0 + ti * 16 + 8 * (l >> 4);
      int cb = c0 + tj * 16 + (l & 15);
      for (int v2 = 0; v2 < 8; v2++) {
        int m = mb + v2;                 // global row = b*T + t
        int bidx = m >> 11, t = m & (TT - 1);
        float val = acc[ti * 4 + tj][v2];
        int c = cb;
        if (c < DIM) {                   // ---- q with LRPE
          int hh = c >> 7, j = c & (E - 1);
          float th = theta[hh * E + j] * (float)t;
          float sn, cs;
          __sincosf(th, &sn, &cs);
          size_t base = (((size_t)(bidx * HEADS + hh)) * TT + t) * DK;
          qout[base + j] = f2bf_bits(val * cs);
          qout[base + E + j] = f2bf_bits(val * sn);
        } else if (c < 2 * DIM) {        // ---- k with LRPE
          int cc = c - DIM;
          int hh = cc >> 7, j = cc & (E - 1);
          float th = theta[hh * E + j] * (float)t;
          float sn, cs;
          __sincosf(th, &sn, &cs);
          size_t base = (((size_t)(bidx * HEADS + hh)) * TT + t) * DK;
          kout[base + j] = f2bf_bits(val * cs);
          kout[base + E + j] = f2bf_bits(val * sn);
        } else {                         // ---- v passthrough
          int cc = c - 2 * DIM;
          int hh = cc >> 7, j = cc & (E - 1);
          vout[(((size_t)(bidx * HEADS + hh)) * TT + t) * E + j] = f2bf_bits(val);
        }
      }
    }
  }
}

// ---------------------------------------------------------------- lightning attention
// One block per (b,h). 8 waves. Recurrent stateT (dv=128 x dk=256) lives in LDS,
// per-chunk GEMMs on WMMA. y written head-interleaved as (B,T,2048) bf16.
__global__ __launch_bounds__(256) void lightning_attn_kernel(
    const u16* __restrict__ qg, const u16* __restrict__ kg, const u16* __restrict__ vg,
    u16* __restrict__ yg) {
  __shared__ __align__(32) u16 sStateT[E * DK];      // 64 KB: stateT[n][d]
  __shared__ __align__(32) u16 sKT[DK * CHUNK];      // 64 KB: kT[d][j]
  __shared__ __align__(32) u16 sVT[E * CHUNK];       // 32 KB: vT[n][j]
  __shared__ __align__(32) u16 sVKD[E * CHUNK];      // 32 KB: kdec[j]*vT[n][j]
  __shared__ __align__(32) u16 sS[CHUNK * CHUNK];    // 32 KB: decayed scores
  __shared__ float sQdec[CHUNK];
  __shared__ float sKdec[CHUNK];

  int bh = blockIdx.x;             // 0..63
  int b = bh >> 4, h = bh & 15;
  int tid = threadIdx.x;
  int w = tid >> 5, l = tid & 31;
  int wr = w >> 1, wc = w & 1;
  int r0 = wr * 32, n0 = wc * 64;  // wave tile within 128x128

  float s = exp2f(-0.5f * (float)(h + 1));            // ALiBi slope for h of 16 heads
  float bdecay = __expf(-s * (float)CHUNK);

  for (int i = tid; i < E * DK; i += 256) sStateT[i] = 0;
  if (tid < CHUNK) {
    sQdec[tid] = __expf(-s * ((float)tid + 1.0f));
    sKdec[tid] = __expf(-s * ((float)(CHUNK - 1) - (float)tid));
  }
  __syncthreads();

  const u16* qbh = qg + (size_t)bh * TT * DK;
  const u16* kbh = kg + (size_t)bh * TT * DK;
  const u16* vbh = vg + (size_t)bh * TT * E;

  for (int c0 = 0; c0 < TT; c0 += CHUNK) {
    // ---- stage vT, kdec*vT, kT into LDS
    for (int i = tid; i < E * CHUNK; i += 256) {
      int j = i >> 7, n = i & (E - 1);
      u16 raw = vbh[(size_t)(c0 + j) * E + n];
      sVT[n * CHUNK + j] = raw;
      sVKD[n * CHUNK + j] = f2bf_bits(bf2f(raw) * sKdec[j]);
    }
    for (int i = tid; i < CHUNK * DK; i += 256) {
      int j = i >> 8, d = i & (DK - 1);
      sKT[d * CHUNK + j] = kbh[(size_t)(c0 + j) * DK + d];
    }
    // ---- prefetch next chunk's q/k/v into cache while this chunk computes
    if (c0 + CHUNK < TT) {
      const char* nq = (const char*)(qbh + (size_t)(c0 + CHUNK) * DK);
      const char* nk = (const char*)(kbh + (size_t)(c0 + CHUNK) * DK);
      const char* nv = (const char*)(vbh + (size_t)(c0 + CHUNK) * E);
      for (int i = tid; i < 512; i += 256) {     // 512 x 128B = 64KB each for q,k
        __builtin_prefetch(nq + i * 128, 0, 3);
        __builtin_prefetch(nk + i * 128, 0, 3);
        if (i < 256) __builtin_prefetch(nv + i * 128, 0, 3);   // 32KB
      }
    }
    __syncthreads();

    const u16* qc = qbh + (size_t)c0 * DK;
    const u16* kc = kbh + (size_t)c0 * DK;

    // ---- G1: S = q @ k^T   (128x128, K=256), operands straight from global
    v8f acc[8];
    v8f z = {};
    for (int i = 0; i < 8; i++) acc[i] = z;
    for (int kk = 0; kk < DK; kk += 32) {
      v16bf a0 = frag_ld(qc, DK, r0, kk);
      v16bf a1 = frag_ld(qc, DK, r0 + 16, kk);
      v16bf b0 = frag_ld(kc, DK, n0, kk);
      v16bf b1 = frag_ld(kc, DK, n0 + 16, kk);
      v16bf b2 = frag_ld(kc, DK, n0 + 32, kk);
      v16bf b3 = frag_ld(kc, DK, n0 + 48, kk);
      wmma8(a0, a1, b0, b1, b2, b3, acc);
    }
    // causal intra-chunk decay mask, store bf16 score tile to LDS
    for (int ti = 0; ti < 2; ti++) {
      for (int tj = 0; tj < 4; tj++) {
        int ib = r0 + ti * 16 + 8 * (l >> 4);
        int j = n0 + tj * 16 + (l & 15);
        for (int v2 = 0; v2 < 8; v2++) {
          int i = ib + v2;
          float val = (i >= j) ? acc[ti * 4 + tj][v2] * __expf(-s * (float)(i - j)) : 0.0f;
          sS[i * CHUNK + j] = f2bf_bits(val);
        }
      }
    }
    __syncthreads();

    // ---- G3: acc = q @ stateT^T  (inter-chunk term, K=256)
    for (int i = 0; i < 8; i++) acc[i] = z;
    for (int kk = 0; kk < DK; kk += 32) {
      v16bf a0 = frag_ld(qc, DK, r0, kk);
      v16bf a1 = frag_ld(qc, DK, r0 + 16, kk);
      v16bf b0 = frag_ld(sStateT, DK, n0, kk);
      v16bf b1 = frag_ld(sStateT, DK, n0 + 16, kk);
      v16bf b2 = frag_ld(sStateT, DK, n0 + 32, kk);
      v16bf b3 = frag_ld(sStateT, DK, n0 + 48, kk);
      wmma8(a0, a1, b0, b1, b2, b3, acc);
    }
    // q_decay factors out onto output rows
    for (int ti = 0; ti < 2; ti++) {
      for (int tj = 0; tj < 4; tj++) {
        int ib = r0 + ti * 16 + 8 * (l >> 4);
        for (int v2 = 0; v2 < 8; v2++) acc[ti * 4 + tj][v2] *= sQdec[ib + v2];
      }
    }
    // ---- G2: acc += Sdec @ v  (K=128, A from LDS scores, B from vT)
    for (int kk = 0; kk < CHUNK; kk += 32) {
      v16bf a0 = frag_ld(sS, CHUNK, r0, kk);
      v16bf a1 = frag_ld(sS, CHUNK, r0 + 16, kk);
      v16bf b0 = frag_ld(sVT, CHUNK, n0, kk);
      v16bf b1 = frag_ld(sVT, CHUNK, n0 + 16, kk);
      v16bf b2 = frag_ld(sVT, CHUNK, n0 + 32, kk);
      v16bf b3 = frag_ld(sVT, CHUNK, n0 + 48, kk);
      wmma8(a0, a1, b0, b1, b2, b3, acc);
    }
    // write O chunk, head-interleaved into y (B,T,2048) as bf16
    for (int ti = 0; ti < 2; ti++) {
      for (int tj = 0; tj < 4; tj++) {
        int ib = r0 + ti * 16 + 8 * (l >> 4);
        int nb = n0 + tj * 16 + (l & 15);
        for (int v2 = 0; v2 < 8; v2++) {
          int t = c0 + ib + v2;
          yg[((size_t)(b * TT + t)) * DIM + h * E + nb] = f2bf_bits(acc[ti * 4 + tj][v2]);
        }
      }
    }
    __syncthreads();   // all reads of old stateT done before update

    // ---- G4: stateT = bdecay*stateT + (kdec∘v)^T @ k
    // Wave w owns stateT rows [w*16, w*16+16), all 16 d-tiles. K = chunk j dim.
    {
      int nb4 = w * 16;
      v8f sa[16];
      for (int td = 0; td < 16; td++) {
        int db = td * 16;
        v8f c;
        for (int v2 = 0; v2 < 8; v2++) {
          int n = nb4 + v2 + 8 * (l >> 4);
          int d = db + (l & 15);
          c[v2] = bdecay * bf2f(sStateT[n * DK + d]);
        }
        sa[td] = c;
      }
      for (int kk = 0; kk < CHUNK; kk += 32) {
        v16bf a = frag_ld(sVKD, CHUNK, nb4, kk);    // pre-scaled A fragment
        for (int td = 0; td < 16; td++) {
          v16bf bfrag = frag_ld(sKT, CHUNK, td * 16, kk);
          sa[td] = WMMA_BF16(a, bfrag, sa[td]);
        }
      }
      for (int td = 0; td < 16; td++) {
        int db = td * 16;
        for (int v2 = 0; v2 < 8; v2++) {
          int n = nb4 + v2 + 8 * (l >> 4);
          int d = db + (l & 15);
          sStateT[n * DK + d] = f2bf_bits(sa[td][v2]);
        }
      }
    }
    __syncthreads();   // stateT visible + vT/kT/vkd free for next chunk
  }
}

// ---------------------------------------------------------------- output projection
// out = y(8192x2048) @ w_proj^T(2048x2048), fp32 out.
__global__ __launch_bounds__(256) void proj_kernel(
    const u16* __restrict__ yb, const u16* __restrict__ wb, float* __restrict__ out) {
  int tid = threadIdx.x;
  int w = tid >> 5, l = tid & 31;
  int wr = w >> 1, wc = w & 1;
  int r0 = blockIdx.x * 128 + wr * 32;
  int c0 = blockIdx.y * 128 + wc * 64;

  v8f acc[8];
  v8f z = {};
  for (int i = 0; i < 8; i++) acc[i] = z;

  for (int kk = 0; kk < DIM; kk += 32) {
    v16bf a0 = frag_ld(yb, DIM, r0, kk);
    v16bf a1 = frag_ld(yb, DIM, r0 + 16, kk);
    v16bf b0 = frag_ld(wb, DIM, c0, kk);
    v16bf b1 = frag_ld(wb, DIM, c0 + 16, kk);
    v16bf b2 = frag_ld(wb, DIM, c0 + 32, kk);
    v16bf b3 = frag_ld(wb, DIM, c0 + 48, kk);
    wmma8(a0, a1, b0, b1, b2, b3, acc);
  }
  for (int ti = 0; ti < 2; ti++) {
    for (int tj = 0; tj < 4; tj++) {
      int mb = r0 + ti * 16 + 8 * (l >> 4);
      int cb = c0 + tj * 16 + (l & 15);
      for (int v2 = 0; v2 < 8; v2++) {
        out[(size_t)(mb + v2) * DIM + cb] = acc[ti * 4 + tj][v2];
      }
    }
  }
}

// ---------------------------------------------------------------- launch
extern "C" void kernel_launch(void* const* d_in, const int* in_sizes, int n_in,
                              void* d_out, int out_size, void* d_ws, size_t ws_size,
                              hipStream_t stream) {
  (void)in_sizes; (void)n_in; (void)out_size; (void)ws_size;
  const float* x     = (const float*)d_in[0];   // (4,2048,2048)
  const float* wqkv  = (const float*)d_in[1];   // (6144,2048)
  const float* wproj = (const float*)d_in[2];   // (2048,2048)
  const float* theta = (const float*)d_in[3];   // (16,1,128)
  float* out = (float*)d_out;                   // (4,2048,2048) fp32

  char* ws = (char*)d_ws;
  size_t off = 0;
  u16* xb    = (u16*)(ws + off); off += (size_t)BB * TT * DIM * 2;          // 32 MB
  u16* wqkvb = (u16*)(ws + off); off += (size_t)3 * DIM * DIM * 2;          // 24 MB
  u16* wprjb = (u16*)(ws + off); off += (size_t)DIM * DIM * 2;              //  8 MB
  u16* qb    = (u16*)(ws + off); off += (size_t)BB * HEADS * TT * DK * 2;   // 64 MB
  u16* kb    = (u16*)(ws + off); off += (size_t)BB * HEADS * TT * DK * 2;   // 64 MB
  u16* vb    = (u16*)(ws + off); off += (size_t)BB * HEADS * TT * E * 2;    // 32 MB
  u16* yb    = (u16*)(ws + off); off += (size_t)BB * TT * DIM * 2;          // 32 MB

  long nx = (long)BB * TT * DIM / 4;
  long nwq = (long)3 * DIM * DIM / 4;
  long nwp = (long)DIM * DIM / 4;
  cast_bf16_kernel<<<2048, 256, 0, stream>>>(x, xb, nx);
  cast_bf16_kernel<<<2048, 256, 0, stream>>>(wqkv, wqkvb, nwq);
  cast_bf16_kernel<<<2048, 256, 0, stream>>>(wproj, wprjb, nwp);

  dim3 gq((BB * TT) / 128, (3 * DIM) / 128);        // 64 x 48
  qkv_lrpe_kernel<<<gq, 256, 0, stream>>>(xb, wqkvb, theta, qb, kb, vb);

  lightning_attn_kernel<<<BB * HEADS, 256, 0, stream>>>(qb, kb, vb, yb);

  dim3 gp((BB * TT) / 128, DIM / 128);              // 64 x 16
  proj_kernel<<<gp, 256, 0, stream>>>(yb, wprjb, out);
}
